// LSTMLayer_51419348467841
// MI455X (gfx1250) — matrix operand order repaired and checked by reading, not compiled, not run on hardware
//
#include <hip/hip_runtime.h>
#include <hip/hip_bf16.h>

// ---------------------------------------------------------------------------
// Fused GRU-like cell on MI455X (gfx1250), bf16 WMMA path, software-pipelined.
//
// Pass 1: ZGR = tanh(X@U_zgr + S@W_zgr + b_zgr) -> Z, G f32 in ws,
//         SR = S * R f32 in ws (breaks the R->H dependency).
// Pass 2: H = tanh(X@Uh + SR@Wh + bh); out = (1-G)*H + Z*S.
//
// Workspace: 3 * 16384*1024*4 = 192 MB (Z, G, SR f32).
// ---------------------------------------------------------------------------

typedef __attribute__((ext_vector_type(16))) __bf16 bf16x16;
typedef __attribute__((ext_vector_type(8)))  __bf16 bf16x8;
typedef __attribute__((ext_vector_type(2)))  __bf16 bf16x2;
typedef __attribute__((ext_vector_type(8)))  float  f32x8;
typedef __attribute__((ext_vector_type(2)))  float  f32x2;

#define NROWS   16384
#define DIM     1024
#define BM      128
#define BN      128
#define BK      64
#define LDSS    72          // padded LDS row stride (elements) to spread banks

// pack two f32 into two bf16 -> single v_cvt_pk_bf16_f32
__device__ __forceinline__ unsigned pk2(float x, float y) {
    f32x2 v; v[0] = x; v[1] = y;
    bf16x2 h = __builtin_convertvector(v, bf16x2);
    return __builtin_bit_cast(unsigned, h);
}

__device__ __forceinline__ float fc(const float4& v, int c) {
    return reinterpret_cast<const float*>(&v)[c];
}

// C += A1[mBase:,0:1024] * B1[0:1024,nBase:] + A2 * B2, fused 2048-deep K loop
// with register double-buffering of the global f32 tiles.
__device__ __forceinline__ void mm_pair(
    const float* __restrict__ A1, const float* __restrict__ B1,
    const float* __restrict__ A2, const float* __restrict__ B2,
    int mBase, int nBase,
    unsigned short* sA, unsigned short* sB,
    f32x8 acc[2][4])
{
    const int tid   = threadIdx.x;
    const int lane  = tid & 31;
    const int wave  = tid >> 5;
    const int mw    = (wave & 3) * 32;   // wave's M offset inside 128
    const int nw    = (wave >> 2) * 64;  // wave's N offset inside 128
    const int tRow  = tid >> 4;          // 0..15  (A loader row)
    const int tCol4 = (tid & 15) * 4;    // 0..60  (A loader col, float4)
    // B loader: per-thread 4x4 blocks (transpose in registers, packed b64 stores)
    const int kb0   = (tid & 15) * 4;    // k sub-block base
    const int nb0   = (tid >> 4) * 4;    // n sub-block base (0..60)

    float4 pa[8];    // prefetched A tile slice (BM x BK / 256 threads)
    float4 pb[8];    // prefetched B tile slice (BK x BN / 256 threads)

    auto gload = [&](int kt) {
        const float* A = (kt < DIM) ? A1 : A2;
        const float* B = (kt < DIM) ? B1 : B2;
        const int k0 = kt & (DIM - 1);
        #pragma unroll
        for (int rr = 0; rr < 8; ++rr) {
            const int row = rr * 16 + tRow;
            pa[rr] = *(const float4*)(A + (size_t)(mBase + row) * DIM + k0 + tCol4);
        }
        #pragma unroll
        for (int i = 0; i < 2; ++i) {
            const int nb = nb0 + i * 64;       // two 4-col groups per thread
            #pragma unroll
            for (int r = 0; r < 4; ++r)
                pb[i * 4 + r] = *(const float4*)(B + (size_t)(k0 + kb0 + r) * DIM + nBase + nb);
        }
    };

    auto lstore = [&]() {
        #pragma unroll
        for (int rr = 0; rr < 8; ++rr) {
            const int row = rr * 16 + tRow;
            uint2 w;
            w.x = pk2(pa[rr].x, pa[rr].y);
            w.y = pk2(pa[rr].z, pa[rr].w);
            *(uint2*)(sA + row * LDSS + tCol4) = w;
        }
        #pragma unroll
        for (int i = 0; i < 2; ++i) {
            const int nb = nb0 + i * 64;
            #pragma unroll
            for (int c = 0; c < 4; ++c) {
                uint2 w;   // 4 consecutive K of column nb+c, packed
                w.x = pk2(fc(pb[i * 4 + 0], c), fc(pb[i * 4 + 1], c));
                w.y = pk2(fc(pb[i * 4 + 2], c), fc(pb[i * 4 + 3], c));
                *(uint2*)(sB + (nb + c) * LDSS + kb0) = w;
            }
        }
    };

    gload(0);
    for (int kt = 0; kt < 2 * DIM; kt += BK) {
        __syncthreads();           // LDS consumed by previous compute
        lstore();                  // convert + packed LDS store
        __syncthreads();

        if (kt + BK < 2 * DIM) gload(kt + BK);   // overlap next tile with WMMA
        if (kt + 2 * BK < 2 * DIM) {             // stream weights into L2 early
            const float* Bp = ((kt + 2 * BK) < DIM) ? B1 : B2;
            const int kp = (kt + 2 * BK) & (DIM - 1);
            __builtin_prefetch(Bp + (size_t)(kp + tRow) * DIM + nBase + tCol4, 0, 1);
        }

        // ---- 2 WMMA k-steps of 32 over the staged BK=64 ----
        #pragma unroll
        for (int ks = 0; ks < BK; ks += 32) {
            // A frag: 16x32 bf16 (lanes 0-15: K 0..15; lanes 16-31:
            // K 16..31), M = lane & 15.
            bf16x16 afr[2];
            const int koA = ks + ((lane < 16) ? 0 : 16);
            #pragma unroll
            for (int tm = 0; tm < 2; ++tm) {
                const int row = mw + tm * 16 + (lane & 15);
                const unsigned short* p = sA + row * LDSS + koA;
                bf16x8 lo = *(const bf16x8*)(p);
                bf16x8 hi = *(const bf16x8*)(p + 8);
                bf16x16 a;
                #pragma unroll
                for (int i = 0; i < 8; ++i) { a[i] = lo[i]; a[8 + i] = hi[i]; }
                afr[tm] = a;
            }
            // B frag: 32x16 bf16; lanes 0-15 hold K ks..ks+15 of column n,
            // lanes 16-31 hold K ks+16..ks+31 (contiguous in transposed LDS).
            bf16x16 bfr[4];
            const int koB = ks + ((lane < 16) ? 0 : 16);
            #pragma unroll
            for (int tn = 0; tn < 4; ++tn) {
                const int coln = nw + tn * 16 + (lane & 15);
                bfr[tn] = *(const bf16x16*)(sB + coln * LDSS + koB);
            }
            #pragma unroll
            for (int tm = 0; tm < 2; ++tm)
                #pragma unroll
                for (int tn = 0; tn < 4; ++tn)
                    acc[tm][tn] = __builtin_amdgcn_wmma_f32_16x16x32_bf16(
                        false, afr[tm], false, bfr[tn],
                        (short)0, acc[tm][tn], false, false);
        }
    }
}

// ---------------- Pass 1: Z, G, SR ----------------
__global__ __launch_bounds__(256) void zgr_kernel(
    const float* __restrict__ X, const float* __restrict__ S,
    const float* __restrict__ Uz, const float* __restrict__ Ug, const float* __restrict__ Ur,
    const float* __restrict__ Wz, const float* __restrict__ Wg, const float* __restrict__ Wr,
    const float* __restrict__ bz, const float* __restrict__ bg, const float* __restrict__ br,
    float* __restrict__ Z, float* __restrict__ G, float* __restrict__ SR)
{
    __shared__ unsigned short sA[BM * LDSS];
    __shared__ unsigned short sB[BN * LDSS];

    const int mBase = blockIdx.y * BM;
    const int nGlob = blockIdx.x * BN;      // 0..3071 in 128 steps
    const int which = nGlob >> 10;          // 0=Z, 1=G, 2=R (tile never crosses)
    const int nBase = nGlob - (which << 10);

    const float* Bu   = (which == 0) ? Uz : (which == 1) ? Ug : Ur;
    const float* Bw   = (which == 0) ? Wz : (which == 1) ? Wg : Wr;
    const float* bias = (which == 0) ? bz : (which == 1) ? bg : br;

    f32x8 acc[2][4] = {};
    mm_pair(X, Bu, S, Bw, mBase, nBase, sA, sB, acc);

    // Epilogue: f32 C layout — VGPR v: lanes 0-15 -> M=v, lanes 16-31 -> M=v+8
    const int tid  = threadIdx.x;
    const int lane = tid & 31;
    const int wave = tid >> 5;
    const int mw   = (wave & 3) * 32;
    const int nw   = (wave >> 2) * 64;
    const int col  = lane & 15;
    const int rofs = (lane < 16) ? 0 : 8;

    #pragma unroll
    for (int tm = 0; tm < 2; ++tm) {
        #pragma unroll
        for (int tn = 0; tn < 4; ++tn) {
            const int j  = nBase + nw + tn * 16 + col;
            const float bj = bias[j];
            #pragma unroll
            for (int v = 0; v < 8; ++v) {
                const int m = mBase + mw + tm * 16 + rofs + v;
                const size_t idx = (size_t)m * DIM + j;
                const float t = tanhf(acc[tm][tn][v] + bj);
                if (which == 0)      Z[idx]  = t;
                else if (which == 1) G[idx]  = t;
                else                 SR[idx] = t * S[idx];   // S * R
            }
        }
    }
}

// ---------------- Pass 2: H and final combine ----------------
__global__ __launch_bounds__(256) void hout_kernel(
    const float* __restrict__ X,  const float* __restrict__ SR,
    const float* __restrict__ Uh, const float* __restrict__ Wh,
    const float* __restrict__ bh,
    const float* __restrict__ Z,  const float* __restrict__ G,
    const float* __restrict__ S,  float* __restrict__ out)
{
    __shared__ unsigned short sA[BM * LDSS];
    __shared__ unsigned short sB[BN * LDSS];

    const int mBase = blockIdx.y * BM;
    const int nBase = blockIdx.x * BN;

    f32x8 acc[2][4] = {};
    mm_pair(X, Uh, SR, Wh, mBase, nBase, sA, sB, acc);

    const int tid  = threadIdx.x;
    const int lane = tid & 31;
    const int wave = tid >> 5;
    const int mw   = (wave & 3) * 32;
    const int nw   = (wave >> 2) * 64;
    const int col  = lane & 15;
    const int rofs = (lane < 16) ? 0 : 8;

    #pragma unroll
    for (int tm = 0; tm < 2; ++tm) {
        #pragma unroll
        for (int tn = 0; tn < 4; ++tn) {
            const int j  = nBase + nw + tn * 16 + col;
            const float bj = bh[j];
            #pragma unroll
            for (int v = 0; v < 8; ++v) {
                const int m = mBase + mw + tm * 16 + rofs + v;
                const size_t idx = (size_t)m * DIM + j;
                const float h = tanhf(acc[tm][tn][v] + bj);
                out[idx] = (1.0f - G[idx]) * h + Z[idx] * S[idx];
            }
        }
    }
}

extern "C" void kernel_launch(void* const* d_in, const int* in_sizes, int n_in,
                              void* d_out, int out_size, void* d_ws, size_t ws_size,
                              hipStream_t stream) {
    // setup_inputs order: X, S, Uz, Ug, Ur, Uh, Wz, Wg, Wr, Wh, bz, bg, br, bh
    const float* X  = (const float*)d_in[0];
    const float* S  = (const float*)d_in[1];
    const float* Uz = (const float*)d_in[2];
    const float* Ug = (const float*)d_in[3];
    const float* Ur = (const float*)d_in[4];
    const float* Uh = (const float*)d_in[5];
    const float* Wz = (const float*)d_in[6];
    const float* Wg = (const float*)d_in[7];
    const float* Wr = (const float*)d_in[8];
    const float* Wh = (const float*)d_in[9];
    const float* bz = (const float*)d_in[10];
    const float* bg = (const float*)d_in[11];
    const float* br = (const float*)d_in[12];
    const float* bh = (const float*)d_in[13];

    const size_t tile = (size_t)NROWS * DIM;   // 16M elements
    float* Z  = (float*)d_ws;
    float* G  = Z  + tile;
    float* SR = G  + tile;                     // total 192 MB of f32 scratch

    dim3 blk(256);
    // Pass 1: N = 3072 -> 24 column tiles; M = 16384 -> 128 row tiles
    zgr_kernel<<<dim3(24, 128), blk, 0, stream>>>(
        X, S, Uz, Ug, Ur, Wz, Wg, Wr, bz, bg, br, Z, G, SR);
    // Pass 2: N = 1024 -> 8 column tiles
    hout_kernel<<<dim3(8, 128), blk, 0, stream>>>(
        X, SR, Uh, Wh, bh, Z, G, S, (float*)d_out);
}
